// ModuleCorrelation_86517821214125
// MI455X (gfx1250) — matrix operand order repaired and checked by reading, not compiled, not run on hardware
//
#include <hip/hip_runtime.h>

typedef __attribute__((ext_vector_type(16))) _Float16 v16h;
typedef __attribute__((ext_vector_type(8)))  _Float16 v8h;
typedef __attribute__((ext_vector_type(8)))  float    v8f;

#define NEIGH   7
#define C_DIM   256
#define H_DIM   160
#define W_DIM   160
#define B_DIM   8
#define TILE_W  16
#define KPAD    264   // 256 + 8 f16 pad -> row stride 528 B (multiple of 16 B, bank-spread)

// D(16x32) = A(16x256) x B(256x32) per (b, h, w-tile, di); out diagonals D[m, m+dj].
__global__ __launch_bounds__(64) void corr_wmma_kernel(
    const float* __restrict__ one, const float* __restrict__ two,
    float* __restrict__ out)
{
    __shared__ __align__(32) _Float16 Al[TILE_W * KPAD];  // A[m][c]
    __shared__ __align__(32) _Float16 Bl[32 * KPAD];      // B[n][c] (cols 0..21 valid)
    __shared__ __align__(16) float    Dl[TILE_W * 32];

    const int w0 = blockIdx.x * TILE_W;
    const int h  = blockIdx.y;
    const int b  = blockIdx.z;
    const int t  = threadIdx.x;
    const int lane   = t & 31;
    const int wid    = t >> 5;       // 2 waves: wave handles N-cols wid*16..wid*16+15
    const int Mn     = lane & 15;    // A: row M / B: col N within tile
    const int hiHalf = lane >> 4;    // lane-half selects K sub-range per ISA layout

    // ---- stage A tile: one[b, 0:256, h, w0:w0+16] -> f16 LDS (coalesced 64B rows per c)
    const size_t oneBase = (size_t)b * C_DIM * H_DIM * W_DIM + (size_t)h * W_DIM + w0;
    for (int i = t; i < C_DIM * TILE_W; i += 64) {
        const int c = i >> 4, m = i & 15;
        Al[m * KPAD + c] = (_Float16)one[oneBase + (size_t)c * (H_DIM * W_DIM) + m];
    }
    __syncthreads();

    const size_t twoBase = (size_t)b * C_DIM * H_DIM * W_DIM;
    const float  scale   = 1.0f / (float)C_DIM;

    for (int di = 0; di < NEIGH; ++di) {
        const int  h2    = h + di - 3;
        const bool rowOK = (h2 >= 0) && (h2 < H_DIM);

        // ---- stage B tile: two[b, 0:256, h2, w0-3 .. w0+18], zero-padded OOB
        for (int i = t; i < C_DIM * 22; i += 64) {
            const int c  = i / 22, n = i % 22;
            const int w2 = w0 - 3 + n;
            float v = 0.0f;
            if (rowOK && (unsigned)w2 < (unsigned)W_DIM)
                v = two[twoBase + (size_t)c * (H_DIM * W_DIM) + (size_t)h2 * W_DIM + w2];
            Bl[n * KPAD + c] = (_Float16)v;
        }
        __syncthreads();

        // ---- K-chain of v_wmma_f32_16x16x32_f16, fp32 accumulate
        v8f acc = {};
        #pragma unroll
        for (int kk = 0; kk < 8; ++kk) {
            // A fragment: lanes 0-15 hold K{+0..7,+16..23}, lanes 16-31 K{+8..15,+24..31}
            const _Float16* ap = &Al[Mn * KPAD + kk * 32 + hiHalf * 8];
            const v8h alo = *(const v8h*)ap;
            const v8h ahi = *(const v8h*)(ap + 16);
            // B fragment: lanes 0-15 hold K{+0..7,+16..23}, lanes 16-31 K{+8..15,+24..31}, N = lane%16
            const _Float16* bp = &Bl[(wid * 16 + Mn) * KPAD + kk * 32 + hiHalf * 8];
            const v8h blo = *(const v8h*)bp;
            const v8h bhi = *(const v8h*)(bp + 16);
            v16h afrag, bfrag;
            #pragma unroll
            for (int q = 0; q < 8; ++q) {
                afrag[q] = alo[q]; afrag[8 + q] = ahi[q];
                bfrag[q] = blo[q]; bfrag[8 + q] = bhi[q];
            }
            acc = __builtin_amdgcn_wmma_f32_16x16x32_f16(
                false, afrag, false, bfrag, (short)0, acc, false, false);
        }

        // ---- spill D: VGPR r -> (M = r + 8*hiHalf, N = wid*16 + Mn)
        #pragma unroll
        for (int r = 0; r < 8; ++r)
            Dl[(r + 8 * hiHalf) * 32 + wid * 16 + Mn] = acc[r];
        __syncthreads();

        // ---- gather diagonals: out[b, di*7+dj, h, w0+m] = D[m, m+dj] / C
        for (int idx = t; idx < TILE_W * NEIGH; idx += 64) {
            const int m = idx / NEIGH, dj = idx % NEIGH;
            const float v = Dl[m * 32 + m + dj] * scale;
            out[(((size_t)b * (NEIGH * NEIGH) + di * NEIGH + dj) * H_DIM + h) * W_DIM
                + w0 + m] = v;
        }
        __syncthreads();  // before next di overwrites Bl / Dl
    }
}

extern "C" void kernel_launch(void* const* d_in, const int* in_sizes, int n_in,
                              void* d_out, int out_size, void* d_ws, size_t ws_size,
                              hipStream_t stream) {
    const float* one = (const float*)d_in[0];
    const float* two = (const float*)d_in[1];
    // d_in[2] = intStride (always 1 in this workload; stride-1 path implemented)
    float* out = (float*)d_out;
    (void)in_sizes; (void)n_in; (void)out_size; (void)d_ws; (void)ws_size;

    dim3 grid(W_DIM / TILE_W, H_DIM, B_DIM);   // 10 x 160 x 8 = 12,800 blocks
    corr_wmma_kernel<<<grid, 64, 0, stream>>>(one, two, out);
}
